// GCN_layer_60773787238589
// MI455X (gfx1250) — matrix operand order repaired and checked by reading, not compiled, not run on hardware
//
#include <hip/hip_runtime.h>

// ---------------- problem constants ----------------
#define NN   8192
#define DIM  256
#define H1D  128
#define BN   64      // key block per iteration
#define BNP  72      // padded LDS row stride (halves): 144B, 16B aligned, conflict-free
#define WAVES 4      // waves per block in the WMMA kernels

typedef __attribute__((ext_vector_type(16))) _Float16 v16h;
typedef __attribute__((ext_vector_type(8)))  _Float16 v8h;
typedef __attribute__((ext_vector_type(8)))  float    v8f;

// ---------------- workspace layout (bytes) ----------------
#define OFF_XH  (size_t)0                       // 8192*256 f16  = 4 MiB
#define OFF_XT  (OFF_XH + (size_t)NN*DIM*2)     // 256*8192 f16  = 4 MiB
#define OFF_RM  (OFF_XT + (size_t)DIM*NN*2)     // 8192 f32
#define OFF_RS  (OFF_RM + (size_t)NN*4)         // 8192 f32
#define OFF_O1  (OFF_RS + (size_t)NN*4)         // 8192*256 f32 = 8 MiB
#define OFF_H   (OFF_O1 + (size_t)NN*DIM*4)     // 8192*128 f16 = 2 MiB
#define OFF_HT  (OFF_H  + (size_t)NN*H1D*2)     // 128*8192 f16 = 2 MiB
#define OFF_O2  (OFF_HT + (size_t)H1D*NN*2)     // 8192*128 f32 = 4 MiB

// ---------------- WMMA helpers ----------------
__device__ __forceinline__ v8f wmma_f16(v16h a, v16h b, v8f c) {
    // D = A(16x32 f16) * B(32x16 f16) + C(16x16 f32)
    return __builtin_amdgcn_wmma_f32_16x16x32_f16(
        /*neg_a=*/false, a, /*neg_b=*/false, b,
        /*c_mod=*/(short)0, c, /*reuse_a=*/false, /*reuse_b=*/false);
}

// A fragment (16x32, MxK) from row-major matrix, rows m0..m0+15, cols kBase..kBase+31.
// ISA layout: lanes 0-15 row M=lane, K={0..7,16..23}; lanes 16-31 row M=lane-16, K={8..15,24..31}.
__device__ __forceinline__ v16h load_a_frag(const _Float16* base, int ld,
                                            int m0, int kBase, int lane16, int hi) {
    const _Float16* p = base + (size_t)(m0 + lane16) * ld + kBase + hi * 8;
    v8h lo = *(const v8h*)(p);
    v8h up = *(const v8h*)(p + 16);
    v16h a;
    #pragma unroll
    for (int i = 0; i < 8; ++i) { a[i] = lo[i]; a[8 + i] = up[i]; }
    return a;
}

// B fragment (32x16, KxN) where B[k][n] = rowmajor[n0+n][kBase+k].
// ISA layout: lane = column n; lanes 0-15 hold K={0..7,16..23}, lanes 16-31 hold K={8..15,24..31}.
__device__ __forceinline__ v16h load_b_frag(const _Float16* base, size_t ld,
                                            int n0, long kBase, int lane16, int hi) {
    const _Float16* p = base + (size_t)(n0 + lane16) * ld + kBase + hi * 8;
    v8h lo = *(const v8h*)(p);
    v8h up = *(const v8h*)(p + 16);
    v16h b;
    #pragma unroll
    for (int i = 0; i < 8; ++i) { b[i] = lo[i]; b[8 + i] = up[i]; }
    return b;
}

// ---------------- kernel 1: f32 -> f16 + transpose ----------------
__global__ void k_cvt(const float* __restrict__ x,
                      _Float16* __restrict__ xh, _Float16* __restrict__ xT) {
    int idx = blockIdx.x * blockDim.x + threadIdx.x;
    if (idx >= NN * DIM) return;
    int i = idx / DIM, c = idx - i * DIM;
    float v = x[idx];
    xh[idx] = (_Float16)v;
    xT[(size_t)c * NN + i] = (_Float16)v;
}

// ---------------- kernel 2: row softmax stats via WMMA + online max/sum ----------------
__global__ __launch_bounds__(32 * WAVES, 1)
void k_stats(const _Float16* __restrict__ xh,
             float* __restrict__ rowmax, float* __restrict__ rowsum) {
    const int lane = threadIdx.x & 31, wave = threadIdx.x >> 5;
    const int lane16 = lane & 15, hi = lane >> 4;
    const int iBase = (blockIdx.x * WAVES + wave) * 16;

    v16h aq[DIM / 32];
    #pragma unroll
    for (int kc = 0; kc < DIM / 32; ++kc)
        aq[kc] = load_a_frag(xh, DIM, iBase, kc * 32, lane16, hi);

    float m[8], s[8];
    #pragma unroll
    for (int r = 0; r < 8; ++r) { m[r] = -3.0e38f; s[r] = 0.0f; }

    for (int jBase = 0; jBase < NN; jBase += BN) {
        v8f acc[BN / 16];
        #pragma unroll
        for (int nt = 0; nt < BN / 16; ++nt) acc[nt] = (v8f)(0.0f);
        #pragma unroll
        for (int kc = 0; kc < DIM / 32; ++kc) {
            #pragma unroll
            for (int nt = 0; nt < BN / 16; ++nt) {
                v16h b = load_b_frag(xh, DIM, jBase + nt * 16, kc * 32, lane16, hi);
                acc[nt] = wmma_f16(aq[kc], b, acc[nt]);
            }
        }
        // online max / sum per output row (C layout: VGPR r -> row r + 8*hi, lanes of a
        // 16-group span N; reduce across the 16-lane group with xor shuffles)
        #pragma unroll
        for (int r = 0; r < 8; ++r) {
            float tm = acc[0][r];
            #pragma unroll
            for (int nt = 1; nt < BN / 16; ++nt) tm = fmaxf(tm, acc[nt][r]);
            #pragma unroll
            for (int off = 1; off < 16; off <<= 1) tm = fmaxf(tm, __shfl_xor(tm, off, 32));
            float nm = fmaxf(m[r], tm);
            float ts = 0.0f;
            #pragma unroll
            for (int nt = 0; nt < BN / 16; ++nt) ts += __expf(acc[nt][r] - nm);
            #pragma unroll
            for (int off = 1; off < 16; off <<= 1) ts += __shfl_xor(ts, off, 32);
            s[r] = s[r] * __expf(m[r] - nm) + ts;
            m[r] = nm;
        }
    }
    if (lane16 == 0) {
        #pragma unroll
        for (int r = 0; r < 8; ++r) {
            int row = iBase + r + hi * 8;
            rowmax[row] = m[r];
            rowsum[row] = s[r];
        }
    }
}

// ---------------- kernels 3 & 5: O = softmax(x x^T) @ V (stats precomputed) ----------------
template <int W>
__global__ __launch_bounds__(32 * WAVES, 1)
void k_av(const _Float16* __restrict__ xh, const _Float16* __restrict__ vT,
          const float* __restrict__ rowmax, const float* __restrict__ rowsum,
          float* __restrict__ O) {
    __shared__ __align__(16) _Float16 pstrip[WAVES][16 * BNP];
    const int lane = threadIdx.x & 31, wave = threadIdx.x >> 5;
    const int lane16 = lane & 15, hi = lane >> 4;
    const int iBase = (blockIdx.x * WAVES + wave) * 16;
    _Float16* ps = pstrip[wave];

    v16h aq[DIM / 32];
    #pragma unroll
    for (int kc = 0; kc < DIM / 32; ++kc)
        aq[kc] = load_a_frag(xh, DIM, iBase, kc * 32, lane16, hi);

    float mr[8], sinv[8];
    #pragma unroll
    for (int r = 0; r < 8; ++r) {
        int row = iBase + r + hi * 8;
        mr[r] = rowmax[row];
        sinv[r] = 1.0f / rowsum[row];
    }

    v8f o[W / 16];
    #pragma unroll
    for (int nt = 0; nt < W / 16; ++nt) o[nt] = (v8f)(0.0f);

    for (int jBase = 0; jBase < NN; jBase += BN) {
        // --- scores ---
        v8f acc[BN / 16];
        #pragma unroll
        for (int nt = 0; nt < BN / 16; ++nt) acc[nt] = (v8f)(0.0f);
        #pragma unroll
        for (int kc = 0; kc < DIM / 32; ++kc) {
            #pragma unroll
            for (int nt = 0; nt < BN / 16; ++nt) {
                v16h b = load_b_frag(xh, DIM, jBase + nt * 16, kc * 32, lane16, hi);
                acc[nt] = wmma_f16(aq[kc], b, acc[nt]);
            }
        }
        __syncthreads();   // WAR: previous iteration's P reads done before overwrite
        // --- probabilities -> LDS strip (C layout -> row-major [16][BNP]) ---
        #pragma unroll
        for (int nt = 0; nt < BN / 16; ++nt) {
            #pragma unroll
            for (int r = 0; r < 8; ++r) {
                float p = __expf(acc[nt][r] - mr[r]) * sinv[r];
                ps[(r + hi * 8) * BNP + nt * 16 + lane16] = (_Float16)p;
            }
        }
        __syncthreads();   // RAW: P visible before A-fragment reads
        // --- O += P @ V  (A from LDS, B contiguous from transposed V) ---
        #pragma unroll
        for (int kc = 0; kc < BN / 32; ++kc) {
            v16h pa = load_a_frag(ps, BNP, 0, kc * 32, lane16, hi);
            #pragma unroll
            for (int nt = 0; nt < W / 16; ++nt) {
                v16h bv = load_b_frag(vT, NN, nt * 16, (long)jBase + kc * 32, lane16, hi);
                o[nt] = wmma_f16(pa, bv, o[nt]);
            }
        }
    }
    // store O (C layout)
    #pragma unroll
    for (int nt = 0; nt < W / 16; ++nt) {
        #pragma unroll
        for (int r = 0; r < 8; ++r) {
            O[(size_t)(iBase + r + hi * 8) * W + nt * 16 + lane16] = o[nt][r];
        }
    }
}

// ---------------- kernel 4: h = relu(O1 @ gcn1), also transposed f16 copy ----------------
__global__ __launch_bounds__(H1D)
void k_h(const float* __restrict__ O1, const float* __restrict__ gcn1,
         _Float16* __restrict__ h, _Float16* __restrict__ hT) {
    __shared__ float row[DIM];
    int i = blockIdx.x;
    int c = threadIdx.x;                  // 128 threads = H1 columns
    for (int k = threadIdx.x; k < DIM; k += H1D) row[k] = O1[(size_t)i * DIM + k];
    __syncthreads();
    float acc = 0.0f;
    #pragma unroll 4
    for (int k = 0; k < DIM; ++k) acc = fmaf(row[k], gcn1[k * H1D + c], acc);
    float r = fmaxf(acc, 0.0f);
    h[(size_t)i * H1D + c] = (_Float16)r;
    hT[(size_t)c * NN + i] = (_Float16)r;
}

// ---------------- kernel 6: out = O2 @ gcn2 ----------------
__global__ void k_out(const float* __restrict__ O2, const float* __restrict__ gcn2,
                      float* __restrict__ out) {
    int idx = blockIdx.x * blockDim.x + threadIdx.x;
    if (idx >= NN * 2) return;
    int i = idx >> 1, c = idx & 1;
    float acc = 0.0f;
    #pragma unroll 4
    for (int k = 0; k < H1D; ++k) acc = fmaf(O2[(size_t)i * H1D + k], gcn2[k * 2 + c], acc);
    out[idx] = acc;
}

// ---------------- launcher ----------------
extern "C" void kernel_launch(void* const* d_in, const int* in_sizes, int n_in,
                              void* d_out, int out_size, void* d_ws, size_t ws_size,
                              hipStream_t stream) {
    (void)in_sizes; (void)n_in; (void)out_size; (void)ws_size;
    const float* x    = (const float*)d_in[0];
    // d_in[1] = adj, ignored (reference recomputes it from x)
    const float* gcn1 = (const float*)d_in[2];
    const float* gcn2 = (const float*)d_in[3];
    float* out = (float*)d_out;

    char* ws = (char*)d_ws;
    _Float16* xh = (_Float16*)(ws + OFF_XH);
    _Float16* xT = (_Float16*)(ws + OFF_XT);
    float*    rm = (float*)(ws + OFF_RM);
    float*    rs = (float*)(ws + OFF_RS);
    float*    O1 = (float*)(ws + OFF_O1);
    _Float16* h  = (_Float16*)(ws + OFF_H);
    _Float16* hT = (_Float16*)(ws + OFF_HT);
    float*    O2 = (float*)(ws + OFF_O2);

    const int blkWmma = 32 * WAVES;                 // 128 threads = 4 waves
    const int grdWmma = NN / (16 * WAVES);          // 128 blocks (each wave: 16 rows)

    k_cvt<<<(NN * DIM + 255) / 256, 256, 0, stream>>>(x, xh, xT);
    k_stats<<<grdWmma, blkWmma, 0, stream>>>(xh, rm, rs);
    k_av<DIM><<<grdWmma, blkWmma, 0, stream>>>(xh, xT, rm, rs, O1);   // O1 = adj @ x
    k_h<<<NN, H1D, 0, stream>>>(O1, gcn1, h, hT);                     // h = relu(O1 @ gcn1)
    k_av<H1D><<<grdWmma, blkWmma, 0, stream>>>(xh, hT, rm, rs, O2);   // O2 = adj @ h
    k_out<<<(NN * 2 + 127) / 128, 128, 0, stream>>>(O2, gcn2, out);   // out = O2 @ gcn2
}
